// MyHANAtt3_19481971655182
// MI455X (gfx1250) — matrix operand order repaired and checked by reading, not compiled, not run on hardware
//
#include <hip/hip_runtime.h>
#include <hip/hip_bf16.h>
#include <math.h>

#define HID   128
#define HEADS 8
#define HD    1024
#define NSYM  390
#define NSYMP 416   // NSYM padded to multiple of 32 (K-dim padding)
#define NHERB 811
#define NATTR 35
#define NATTRP 64   // NATTR padded to multiple of 32 (K-dim padding)
#define NALL  1236
#define NB    128
#define EH    32768
#define ES    16384

typedef __attribute__((ext_vector_type(16))) _Float16 v16h;
typedef __attribute__((ext_vector_type(8)))  float    v8f;

static inline unsigned ceil_div(int n, int b) { return (unsigned)((n + b - 1) / b); }

// ---------- device helpers ----------
__device__ __forceinline__ float lrelu02(float v) { return v > 0.f ? v : 0.2f * v; }
__device__ __forceinline__ unsigned fkey(float f) {
  unsigned u = __float_as_uint(f);
  return (u & 0x80000000u) ? ~u : (u | 0x80000000u);
}
__device__ __forceinline__ float funkey(unsigned k) {
  unsigned u = (k & 0x80000000u) ? (k & 0x7FFFFFFFu) : ~k;
  return __uint_as_float(u);
}
__device__ __forceinline__ bool finitef(float f) {
  return ((__float_as_uint(f) >> 23) & 0xFFu) != 0xFFu;
}
__device__ __forceinline__ void cvt16(v16h& v, float4 a, float4 b, float4 c, float4 d) {
  v[0]  = (_Float16)a.x; v[1]  = (_Float16)a.y; v[2]  = (_Float16)a.z; v[3]  = (_Float16)a.w;
  v[4]  = (_Float16)b.x; v[5]  = (_Float16)b.y; v[6]  = (_Float16)b.z; v[7]  = (_Float16)b.w;
  v[8]  = (_Float16)c.x; v[9]  = (_Float16)c.y; v[10] = (_Float16)c.z; v[11] = (_Float16)c.w;
  v[12] = (_Float16)d.x; v[13] = (_Float16)d.y; v[14] = (_Float16)d.z; v[15] = (_Float16)d.w;
}

// ---------- WMMA GEMM: C[M,N] = A[M,K] @ (TRANSB ? B[N,K]^T : B[K,N]) + bias ----------
// REQUIRES: K % 32 == 0. One wave computes a 16x32 tile (two N-subtiles share the A
// fragment -> 2 back-to-back WMMAs per K-step). Branchless, vectorized loads; rows/cols
// are clamped on load (duplicates), only valid C entries stored. KC!=0 = compile-time K.
template <int TRANSB, int KC>
__global__ __launch_bounds__(32)
void gemm_wmma(const float* __restrict__ A, const float* __restrict__ B,
               const float* __restrict__ bias, float* __restrict__ C,
               int M, int N, int Krt)
{
  const int K = KC ? KC : Krt;
  const int lane = threadIdx.x & 31;
  const int half = lane >> 4;
  const int l15  = lane & 15;
  const int tm = blockIdx.y * 16;
  const int tn = blockIdx.x * 32;
  const int arow  = min(tm + l15, M - 1);
  const int bcol0 = min(tn + l15, N - 1);
  const int bcol1 = min(tn + 16 + l15, N - 1);
  // A lane layout (16-bit A 16x32): row = l&15; elem e: K=k0+half*16+e
  const float* Arow = A + (size_t)arow * K + half * 16;
  v8f acc0 = {}, acc1 = {};
  for (int k0 = 0; k0 < K; k0 += 32) {
    v16h av, bv0, bv1;
    cvt16(av,
          *(const float4*)(Arow + k0),      *(const float4*)(Arow + k0 + 4),
          *(const float4*)(Arow + k0 + 8),  *(const float4*)(Arow + k0 + 12));
    // B lane layout (16-bit B 32x16): col = l&15; elem e: K=k0+half*16+e
    if (TRANSB) {
      const float* Bp0 = B + (size_t)bcol0 * K + k0 + half * 16;
      const float* Bp1 = B + (size_t)bcol1 * K + k0 + half * 16;
      cvt16(bv0, *(const float4*)(Bp0),     *(const float4*)(Bp0 + 4),
                 *(const float4*)(Bp0 + 8), *(const float4*)(Bp0 + 12));
      cvt16(bv1, *(const float4*)(Bp1),     *(const float4*)(Bp1 + 4),
                 *(const float4*)(Bp1 + 8), *(const float4*)(Bp1 + 12));
    } else {
      const float* Bp = B + (size_t)(k0 + half * 16) * N;
#pragma unroll
      for (int e = 0; e < 16; ++e) {
        bv0[e] = (_Float16)Bp[(size_t)e * N + bcol0];
        bv1[e] = (_Float16)Bp[(size_t)e * N + bcol1];
      }
    }
    acc0 = __builtin_amdgcn_wmma_f32_16x16x32_f16(false, av, false, bv0,
                                                  (short)0, acc0, false, false);
    acc1 = __builtin_amdgcn_wmma_f32_16x16x32_f16(false, av, false, bv1,
                                                  (short)0, acc1, false, false);
  }
  // C/D layout: col = l&15; VGPR r -> row (l>>4)*8 + r
#pragma unroll
  for (int r = 0; r < 8; ++r) {
    int row = tm + half * 8 + r;
    if (row < M) {
      int c0 = tn + l15, c1 = tn + 16 + l15;
      if (c0 < N) {
        float v = acc0[r];
        if (bias) v += bias[c0];
        C[(size_t)row * N + c0] = v;
      }
      if (c1 < N) {
        float v = acc1[r];
        if (bias) v += bias[c1];
        C[(size_t)row * N + c1] = v;
      }
    }
  }
}

// ---------- small utility kernels ----------
__global__ void fill_f32(float* p, float v, int n) {
  int i = blockIdx.x * blockDim.x + threadIdx.x;
  if (i < n) p[i] = v;
}
__global__ void fill_u32(unsigned* p, unsigned v, int n) {
  int i = blockIdx.x * blockDim.x + threadIdx.x;
  if (i < n) p[i] = v;
}
__global__ void copy_f32(const float* __restrict__ a, float* __restrict__ b, int n) {
  int i = blockIdx.x * blockDim.x + threadIdx.x;
  if (i < n) b[i] = a[i];
}
__global__ void gather_rows(const float* __restrict__ emb, const int* __restrict__ idx,
                            float* __restrict__ out, int n) {
  int t = blockIdx.x * blockDim.x + threadIdx.x;
  if (t >= n * HID) return;
  int i = t / HID, d = t % HID;
  out[t] = emb[(size_t)idx[i] * HID + d];
}
// scatter one-hot ints into K-padded float rows: out[i*NATTRP + k] = kg[i*NATTR + k]
__global__ void kg_pad(const int* __restrict__ kg, float* __restrict__ out, int n) {
  int idx = blockIdx.x * blockDim.x + threadIdx.x;
  if (idx >= n) return;
  int i = idx / NATTR, k = idx % NATTR;
  out[(size_t)i * NATTRP + k] = (float)kg[idx];
}
__global__ void mean3(const float* a, const float* b, const float* c, float* y, int n) {
  int i = blockIdx.x * blockDim.x + threadIdx.x;
  if (i < n) y[i] = (a[i] + b[i] + c[i]) * (1.f / 3.f);
}
__global__ void halfsum(const float* a, const float* b, float* y, int n) {
  int i = blockIdx.x * blockDim.x + threadIdx.x;
  if (i < n) y[i] = (a[i] + b[i]) * 0.5f;
}

// ---------- GAT pieces ----------
__global__ void el_er_kernel(const float* __restrict__ feat, const float* __restrict__ al,
                             const float* __restrict__ ar, float* el, float* er, int n) {
  int idx = blockIdx.x * blockDim.x + threadIdx.x;
  if (idx >= n * HEADS) return;
  int i = idx >> 3, h = idx & 7;
  const float* f = feat + (size_t)i * HD + h * HID;
  const float* a = al + h * HID;
  const float* b = ar + h * HID;
  float sl = 0.f, sr = 0.f;
  for (int d = 0; d < HID; ++d) { float v = f[d]; sl += v * a[d]; sr += v * b[d]; }
  el[idx] = sl; er[idx] = sr;
}
__global__ void edge_pass1(const int* __restrict__ src, const int* __restrict__ dst,
                           const float* __restrict__ el, const float* __restrict__ er,
                           float* ev, unsigned* mkey, int E) {
  int idx = blockIdx.x * blockDim.x + threadIdx.x;
  if (idx >= E * HEADS) return;
  int e = idx >> 3, h = idx & 7;
  int s = src[e], d = dst[e];
  float v = lrelu02(el[s * HEADS + h] + er[d * HEADS + h]);
  ev[idx] = v;
  atomicMax(&mkey[d * HEADS + h], fkey(v));
}
__global__ void edge_pass2(const int* __restrict__ dst, float* ev,
                           const unsigned* __restrict__ mkey, float* ssum, int E) {
  int idx = blockIdx.x * blockDim.x + threadIdx.x;
  if (idx >= E * HEADS) return;
  int e = idx >> 3, h = idx & 7;
  int d = dst[e];
  float m = funkey(mkey[d * HEADS + h]);
  if (!finitef(m)) m = 0.f;
  float ee = __expf(ev[idx] - m);
  ev[idx] = ee;
  atomicAdd(&ssum[d * HEADS + h], ee);
}
__global__ void edge_pass3(const int* __restrict__ src, const int* __restrict__ dst,
                           const float* __restrict__ ev, const float* __restrict__ ssum,
                           const float* __restrict__ feat, float* out, int E) {
  int idx = blockIdx.x * blockDim.x + threadIdx.x;
  if (idx >= E * HEADS) return;
  int e = idx >> 3, h = idx & 7;
  int s = src[e], d = dst[e];
  float alpha = ev[idx] / fmaxf(ssum[d * HEADS + h], 1e-38f);
  const float* fs = feat + (size_t)s * HD + h * HID;
  float* od = out + (size_t)d * HD + h * HID;
  for (int t = 0; t < HID; ++t) atomicAdd(&od[t], alpha * fs[t]);
}
__global__ void bias_elu(float* x, const float* __restrict__ bias, int n) {
  int idx = blockIdx.x * blockDim.x + threadIdx.x;
  if (idx >= n * HD) return;
  float v = x[idx] + bias[idx % HD];
  x[idx] = v > 0.f ? v : (__expf(v) - 1.f);
}

// ---------- HAN semantic attention ----------
__global__ void score_w(const float* __restrict__ T0, const float* __restrict__ T1,
                        const float* __restrict__ sw2, float* wv, int n) {
  int idx = blockIdx.x * blockDim.x + threadIdx.x;
  if (idx >= n * 2) return;
  int i = idx >> 1, m = idx & 1;
  const float* T = (m ? T1 : T0) + (size_t)i * HID;
  float acc = 0.f;
  for (int d = 0; d < HID; ++d) acc += tanhf(T[d]) * sw2[d];
  wv[idx] = acc;
}
__global__ void beta_reduce(const float* __restrict__ wv, float* beta, int n) {
  __shared__ float s0[256], s1[256];
  int t = threadIdx.x;
  float a0 = 0.f, a1 = 0.f;
  for (int i = t; i < n; i += 256) { a0 += wv[2 * i]; a1 += wv[2 * i + 1]; }
  s0[t] = a0; s1[t] = a1; __syncthreads();
  for (int o = 128; o > 0; o >>= 1) {
    if (t < o) { s0[t] += s0[t + o]; s1[t] += s1[t + o]; }
    __syncthreads();
  }
  if (t == 0) {
    float m0 = s0[0] / n, m1 = s1[0] / n;
    float mx = fmaxf(m0, m1);
    float e0 = __expf(m0 - mx), e1 = __expf(m1 - mx);
    float inv = 1.f / (e0 + e1);
    beta[0] = e0 * inv; beta[1] = e1 * inv;
  }
}
__global__ void combine_beta(const float* __restrict__ g0, const float* __restrict__ g1,
                             const float* __restrict__ beta, float* zc, int n) {
  int i = blockIdx.x * blockDim.x + threadIdx.x;
  if (i < n) zc[i] = beta[0] * g0[i] + beta[1] * g1[i];
}
__global__ void tanh_mean_heads(const float* __restrict__ g, float* y, int n) {
  int idx = blockIdx.x * blockDim.x + threadIdx.x;
  if (idx >= n * HID) return;
  int i = idx / HID, d = idx % HID;
  float acc = 0.f;
  for (int h = 0; h < HEADS; ++h) acc += tanhf(g[(size_t)i * HD + h * HID + d]);
  y[idx] = acc * (1.f / HEADS);
}

// ---------- BatchNorm over rows (axis 0) fused with activation ----------
// one block per column j; act: 0=none 1=tanh 2=relu
__global__ void bn_act(const float* __restrict__ x, const float* __restrict__ g,
                       const float* __restrict__ bta, float* y, int M, int act) {
  __shared__ float r1[256], r2[256];
  int j = blockIdx.x, t = threadIdx.x;
  float s = 0.f, s2 = 0.f;
  for (int i = t; i < M; i += 256) { float v = x[(size_t)i * HID + j]; s += v; s2 += v * v; }
  r1[t] = s; r2[t] = s2; __syncthreads();
  for (int o = 128; o > 0; o >>= 1) {
    if (t < o) { r1[t] += r1[t + o]; r2[t] += r2[t + o]; }
    __syncthreads();
  }
  float mu = r1[0] / M;
  float var = r2[0] / M - mu * mu;
  float rs = rsqrtf(var + 1e-5f);
  float gg = g[j], bb = bta[j];
  for (int i = t; i < M; i += 256) {
    float v = gg * (x[(size_t)i * HID + j] - mu) * rs + bb;
    if (act == 1) v = tanhf(v);
    else if (act == 2) v = fmaxf(v, 0.f);
    y[(size_t)i * HID + j] = v;
  }
}

// ---------- attr attention ----------
__global__ void mask_softmax35(float* sc, const int* __restrict__ kg, int n) {
  int i = blockIdx.x * blockDim.x + threadIdx.x;
  if (i >= n) return;
  float mx = -3.0e38f; int any = 0;
  for (int k = 0; k < NATTR; ++k)
    if (kg[i * NATTR + k] > 0) { float v = sc[i * NATTR + k]; mx = v > mx ? v : mx; any = 1; }
  float Z = 0.f;
  for (int k = 0; k < NATTR; ++k) {
    float t = 0.f;
    if (kg[i * NATTR + k] > 0) { t = __expf(sc[i * NATTR + k] - mx); Z += t; }
    sc[i * NATTR + k] = t;
  }
  float inv = any ? 1.f / Z : 0.f;
  for (int k = 0; k < NATTR; ++k) sc[i * NATTR + k] *= inv;
}
__global__ void attri_kernel(const float* __restrict__ sc, const float* __restrict__ attrF,
                             float* out, int n) {
  int idx = blockIdx.x * blockDim.x + threadIdx.x;
  if (idx >= n * HID) return;
  int i = idx / HID, d = idx % HID;
  float acc = 0.f;
  for (int k = 0; k < NATTR; ++k) acc += sc[i * NATTR + k] * attrF[k * HID + d];
  out[idx] = acc;
}

// ---------- batched masked softmax -> e_syn, without materializing (B,390,811) ----------
// writes K-padded rows of length NSYMP (tail zeroed)
__global__ void esyn_kernel(const float* __restrict__ sc2 /*390x811*/,
                            const int* __restrict__ sids /*128x390*/,
                            float* esyn /*128xNSYMP*/) {
  __shared__ float acc[NSYM];
  __shared__ int   msk[NSYM];
  int b = blockIdx.x;
  for (int s = threadIdx.x; s < NSYM; s += blockDim.x) {
    msk[s] = sids[b * NSYM + s];
    acc[s] = 0.f;
  }
  __syncthreads();
  for (int n = threadIdx.x; n < NHERB; n += blockDim.x) {
    float mx = -3.0e38f;
    for (int s = 0; s < NSYM; ++s)
      if (msk[s] > 0) { float v = sc2[(size_t)s * NHERB + n]; mx = v > mx ? v : mx; }
    float Z = 0.f;
    for (int s = 0; s < NSYM; ++s)
      if (msk[s] > 0) Z += __expf(sc2[(size_t)s * NHERB + n] - mx);
    if (Z > 0.f) {
      float inv = 1.f / Z;
      for (int s = 0; s < NSYM; ++s)
        if (msk[s] > 0) atomicAdd(&acc[s], __expf(sc2[(size_t)s * NHERB + n] - mx) * inv);
    }
  }
  __syncthreads();
  for (int s = threadIdx.x; s < NSYMP; s += blockDim.x)
    esyn[b * NSYMP + s] = (s < NSYM) ? acc[s] : 0.f;
}

// ---------- host orchestration ----------
extern "C" void kernel_launch(void* const* d_in, const int* in_sizes, int n_in,
                              void* d_out, int out_size, void* d_ws, size_t ws_size,
                              hipStream_t stream) {
  (void)in_sizes; (void)n_in; (void)out_size; (void)ws_size;

  const float* emb  = (const float*)d_in[0];
  const float* hW   = (const float*)d_in[1];
  const float* hAl  = (const float*)d_in[2];
  const float* hAr  = (const float*)d_in[3];
  const float* hB   = (const float*)d_in[4];
  const float* hSW1 = (const float*)d_in[5];
  const float* hSb1 = (const float*)d_in[6];
  const float* hSW2 = (const float*)d_in[7];
  const float* hPW  = (const float*)d_in[8];
  const float* hPb  = (const float*)d_in[9];
  const float* sW   = (const float*)d_in[10];
  const float* sAl  = (const float*)d_in[11];
  const float* sAr  = (const float*)d_in[12];
  const float* sB   = (const float*)d_in[13];
  const float* sSW1 = (const float*)d_in[14];
  const float* sSb1 = (const float*)d_in[15];
  const float* sSW2 = (const float*)d_in[16];
  const float* sPW  = (const float*)d_in[17];
  const float* sPb  = (const float*)d_in[18];
  const float* hhW  = (const float*)d_in[19];
  const float* hhAl = (const float*)d_in[20];
  const float* hhAr = (const float*)d_in[21];
  const float* hhB  = (const float*)d_in[22];
  const float* ssW  = (const float*)d_in[23];
  const float* ssAl = (const float*)d_in[24];
  const float* ssAr = (const float*)d_in[25];
  const float* ssB  = (const float*)d_in[26];
  const float* Hm1W = (const float*)d_in[27];
  const float* Hm1b = (const float*)d_in[28];
  const float* Hm2W = (const float*)d_in[29];
  const float* Hm2b = (const float*)d_in[30];
  const float* SmW  = (const float*)d_in[31];
  const float* Smb  = (const float*)d_in[32];
  const float* AmW  = (const float*)d_in[33];
  const float* Amb  = (const float*)d_in[34];
  const float* SyW  = (const float*)d_in[35];
  const float* Syb  = (const float*)d_in[36];
  const float* Hbn1g = (const float*)d_in[37];
  const float* Hbn1b = (const float*)d_in[38];
  const float* Hbn2g = (const float*)d_in[39];
  const float* Hbn2b = (const float*)d_in[40];
  const float* Sbng  = (const float*)d_in[41];
  const float* Sbnb  = (const float*)d_in[42];
  const float* Abng  = (const float*)d_in[43];
  const float* Abnb  = (const float*)d_in[44];
  const float* Sybng = (const float*)d_in[45];
  const float* Sybnb = (const float*)d_in[46];
  const int* feature    = (const int*)d_in[47];
  const int* kg         = (const int*)d_in[48];
  const int* sids       = (const int*)d_in[49];
  const int* herb_edges = (const int*)d_in[50];
  const int* sym_edges  = (const int*)d_in[51];
  const int* hh_edges   = (const int*)d_in[52];
  const int* ss_edges   = (const int*)d_in[53];

  float* wsf = (float*)d_ws;
  size_t off = 0;
  auto alloc = [&](size_t nel) -> float* {
    float* p = wsf + off;
    off += (nel + 63) & ~(size_t)63;
    return p;
  };

  float* ALLF  = alloc((size_t)NALL * HID);
  float* FEAT  = alloc((size_t)NHERB * HD);
  float* G0    = alloc((size_t)NHERB * HD);
  float* G1    = alloc((size_t)NHERB * HD);
  float* ZC    = alloc((size_t)NHERB * HD);
  float* EL    = alloc(NHERB * HEADS);
  float* ER    = alloc(NHERB * HEADS);
  float* EV    = alloc((size_t)EH * HEADS);
  unsigned* MKEY = (unsigned*)alloc(NHERB * HEADS);
  float* SSUM  = alloc(NHERB * HEADS);
  float* T0    = alloc((size_t)NHERB * HID);
  float* T1    = alloc((size_t)NHERB * HID);
  float* WV    = alloc(NHERB * 2);
  float* BETA  = alloc(64);
  float* KGF   = alloc((size_t)NHERB * NATTRP);   // K-padded one-hot
  float* ATTR1P = alloc((size_t)NATTRP * HID);    // K-padded attr1 rows
  float* KG1   = alloc((size_t)NHERB * HID);
  float* XH    = alloc((size_t)NHERB * HID);
  float* HERB2 = alloc((size_t)NHERB * HID);
  float* HERB3 = alloc((size_t)NHERB * HID);
  float* SYM2  = alloc((size_t)NSYM * HID);
  float* SYM3  = alloc((size_t)NSYM * HID);
  float* TMPA  = alloc((size_t)NHERB * HID);
  float* TMPB  = alloc((size_t)NHERB * HID);
  float* HERBF = alloc((size_t)NHERB * HID);
  float* SYMF  = alloc((size_t)NSYMP * HID);      // K-padded rows for e_syn GEMM
  float* ATTRF = alloc(NATTR * HID);
  float* SC    = alloc(NHERB * NATTR);
  float* ATTRI = alloc((size_t)NHERB * HID);
  float* SC2   = alloc((size_t)NSYM * NHERB);
  float* ESYN  = alloc((size_t)NB * NSYMP);       // K-padded
  float* ESYN2 = alloc(NB * HID);
  float* TMPC  = alloc(NB * HID);
  float* ESN   = alloc(NB * HID);
  float* HERBF2 = alloc((size_t)NHERB * HID);

  const float* symptom1 = ALLF;
  const float* herb1    = ALLF + (size_t)NSYM * HID;
  const float* attr1    = ALLF + (size_t)(NSYM + NHERB) * HID;

  // dispatch to template instantiations: transB and the known K values compile-time
  auto gemm = [&](const float* A, const float* B, const float* bias, float* C,
                  int M, int N, int K, int tB) {
    dim3 grid(ceil_div(N, 32), ceil_div(M, 16)), blk(32);
    if (tB) {
      if (K == HID) gemm_wmma<1, HID><<<grid, blk, 0, stream>>>(A, B, bias, C, M, N, K);
      else          gemm_wmma<1, 0><<<grid, blk, 0, stream>>>(A, B, bias, C, M, N, K);
    } else {
      switch (K) {
        case HID:    gemm_wmma<0, HID><<<grid, blk, 0, stream>>>(A, B, bias, C, M, N, K); break;
        case HD:     gemm_wmma<0, HD><<<grid, blk, 0, stream>>>(A, B, bias, C, M, N, K); break;
        case NATTRP: gemm_wmma<0, NATTRP><<<grid, blk, 0, stream>>>(A, B, bias, C, M, N, K); break;
        case NSYMP:  gemm_wmma<0, NSYMP><<<grid, blk, 0, stream>>>(A, B, bias, C, M, N, K); break;
        default:     gemm_wmma<0, 0><<<grid, blk, 0, stream>>>(A, B, bias, C, M, N, K); break;
      }
    }
  };

  auto run_gat = [&](const float* x, int n, const int* src, const int* dst, int E,
                     const float* W, const float* al, const float* ar,
                     const float* bias, float* out) {
    gemm(x, W, nullptr, FEAT, n, HD, HID, 0);
    el_er_kernel<<<ceil_div(n * HEADS, 256), 256, 0, stream>>>(FEAT, al, ar, EL, ER, n);
    fill_u32<<<ceil_div(n * HEADS, 256), 256, 0, stream>>>(MKEY, 0u, n * HEADS);
    fill_f32<<<ceil_div(n * HEADS, 256), 256, 0, stream>>>(SSUM, 0.f, n * HEADS);
    fill_f32<<<ceil_div(n * HD, 256), 256, 0, stream>>>(out, 0.f, n * HD);
    edge_pass1<<<ceil_div(E * HEADS, 256), 256, 0, stream>>>(src, dst, EL, ER, EV, MKEY, E);
    edge_pass2<<<ceil_div(E * HEADS, 256), 256, 0, stream>>>(dst, EV, MKEY, SSUM, E);
    edge_pass3<<<ceil_div(E * HEADS, 256), 256, 0, stream>>>(src, dst, EV, SSUM, FEAT, out, E);
    bias_elu<<<ceil_div(n * HD, 256), 256, 0, stream>>>(out, bias, n);
  };

  auto run_han = [&](const float* x, int n, const int* edges, int E,
                     const float* W, const float* al, const float* ar, const float* bias,
                     const float* sw1, const float* sb1, const float* sw2,
                     const float* pw, const float* pb, float* out) {
    for (int m = 0; m < 2; ++m) {
      const int* src = edges + (size_t)m * 2 * E;
      const int* dst = src + E;
      float* G = m ? G1 : G0;
      float* T = m ? T1 : T0;
      run_gat(x, n, src, dst, E, W + (size_t)m * HID * HD,
              al + m * HEADS * HID, ar + m * HEADS * HID, bias + m * HD, G);
      gemm(G, sw1, sb1, T, n, HID, HD, 0);
    }
    score_w<<<ceil_div(n * 2, 256), 256, 0, stream>>>(T0, T1, sw2, WV, n);
    beta_reduce<<<1, 256, 0, stream>>>(WV, BETA, n);
    combine_beta<<<ceil_div(n * HD, 256), 256, 0, stream>>>(G0, G1, BETA, ZC, n * HD);
    gemm(ZC, pw, pb, out, n, HID, HD, 0);
  };

  // 0) embedding gather + K-padded helper buffers
  gather_rows<<<ceil_div(NALL * HID, 256), 256, 0, stream>>>(emb, feature, ALLF, NALL);
  fill_f32<<<ceil_div(NHERB * NATTRP, 256), 256, 0, stream>>>(KGF, 0.f, NHERB * NATTRP);
  kg_pad<<<ceil_div(NHERB * NATTR, 256), 256, 0, stream>>>(kg, KGF, NHERB * NATTR);
  fill_f32<<<ceil_div(NATTRP * HID, 256), 256, 0, stream>>>(ATTR1P, 0.f, NATTRP * HID);
  copy_f32<<<ceil_div(NATTR * HID, 256), 256, 0, stream>>>(attr1, ATTR1P, NATTR * HID);
  fill_f32<<<ceil_div(NSYMP * HID, 256), 256, 0, stream>>>(SYMF, 0.f, NSYMP * HID);

  // 1) HAN for herbs and symptoms
  run_han(herb1, NHERB, herb_edges, EH, hW, hAl, hAr, hB, hSW1, hSb1, hSW2, hPW, hPb, HERB2);
  run_han(symptom1, NSYM, sym_edges, ES, sW, sAl, sAr, sB, sSW1, sSb1, sSW2, sPW, sPb, SYM2);

  // 2) kg1 = kgOneHot @ attr1 (K padded 35->64); xh = (herb1+kg1)*0.5; herb3 = tanh(gat(xh)).mean(h)
  gemm(KGF, ATTR1P, nullptr, KG1, NHERB, HID, NATTRP, 0);
  halfsum<<<ceil_div(NHERB * HID, 256), 256, 0, stream>>>(herb1, KG1, XH, NHERB * HID);
  run_gat(XH, NHERB, hh_edges, hh_edges + EH, EH, hhW, hhAl, hhAr, hhB, G0);
  tanh_mean_heads<<<ceil_div(NHERB * HID, 256), 256, 0, stream>>>(G0, HERB3, NHERB);

  // 3) symptom3 = tanh(gat(symptom1)).mean(heads)
  run_gat(symptom1, NSYM, ss_edges, ss_edges + ES, ES, ssW, ssAl, ssAr, ssB, G0);
  tanh_mean_heads<<<ceil_div(NSYM * HID, 256), 256, 0, stream>>>(G0, SYM3, NSYM);

  // 4) dense + BN + tanh
  mean3<<<ceil_div(NHERB * HID, 256), 256, 0, stream>>>(herb1, HERB2, HERB3, TMPA, NHERB * HID);
  gemm(TMPA, Hm1W, Hm1b, TMPB, NHERB, HID, HID, 0);
  bn_act<<<HID, 256, 0, stream>>>(TMPB, Hbn1g, Hbn1b, HERBF, NHERB, 1);

  mean3<<<ceil_div(NSYM * HID, 256), 256, 0, stream>>>(symptom1, SYM2, SYM3, TMPA, NSYM * HID);
  gemm(TMPA, SmW, Smb, TMPB, NSYM, HID, HID, 0);
  bn_act<<<HID, 256, 0, stream>>>(TMPB, Sbng, Sbnb, SYMF, NSYM, 1);  // rows 390.. stay 0

  gemm(attr1, AmW, Amb, TMPB, NATTR, HID, HID, 0);
  bn_act<<<HID, 256, 0, stream>>>(TMPB, Abng, Abnb, ATTRF, NATTR, 1);

  // 5) herb-attr attention
  gemm(HERBF, ATTRF, nullptr, SC, NHERB, NATTR, HID, 1);
  mask_softmax35<<<ceil_div(NHERB, 256), 256, 0, stream>>>(SC, kg, NHERB);
  attri_kernel<<<ceil_div(NHERB * HID, 256), 256, 0, stream>>>(SC, ATTRF, ATTRI, NHERB);

  // 6) symptom-herb masked softmax -> e_syn (no (B,390,811) materialization)
  gemm(SYMF, HERBF, nullptr, SC2, NSYM, NHERB, HID, 1);
  esyn_kernel<<<NB, 256, 0, stream>>>(SC2, sids, ESYN);
  gemm(ESYN, SYMF, nullptr, ESYN2, NB, HID, NSYMP, 0);  // K padded 390->416

  // 7) final herb update and esn
  halfsum<<<ceil_div(NHERB * HID, 256), 256, 0, stream>>>(HERBF, ATTRI, TMPA, NHERB * HID);
  gemm(TMPA, Hm2W, Hm2b, TMPB, NHERB, HID, HID, 0);
  bn_act<<<HID, 256, 0, stream>>>(TMPB, Hbn2g, Hbn2b, HERBF2, NHERB, 1);

  gemm(ESYN2, SyW, Syb, TMPC, NB, HID, HID, 0);
  bn_act<<<HID, 256, 0, stream>>>(TMPC, Sybng, Sybnb, ESN, NB, 2);

  // 8) output = esn @ herb.T  -> (128, 811)
  gemm(ESN, HERBF2, nullptr, (float*)d_out, NB, NHERB, HID, 1);
}
